// MultiHeadedAttention_6476810682903
// MI455X (gfx1250) — compile-verified, matrix operand order checked by reading, not run on hardware
//
#include <hip/hip_runtime.h>
#include <hip/hip_bf16.h>

// ---------------------------------------------------------------------------
// MultiHeadedAttention for MI455X (gfx1250, wave32, WMMA).
// Pipeline: f32->bf16 convert, 3 projection GEMMs (bf16 in, bf16 out),
// flash-attention with v_wmma_f32_16x16x32_bf16 (+ async LDS staging),
// output GEMM (f32 out).
// ---------------------------------------------------------------------------

#define B_ 4
#define S_ 1024
#define D_ 1024
#define H_ 16
#define DK_ 64
#define M_TOT (B_ * S_)           // 4096 tokens
#define SCALE 0.125f              // 1/sqrt(64)
#define NEGINF (-3.0e38f)

typedef __attribute__((ext_vector_type(16))) __bf16 v16bf;
typedef __attribute__((ext_vector_type(8)))  float  v8f;

union Frag16 { uint4 u4[2]; v16bf v; };

static __device__ inline unsigned short f32_to_bf16(float f) {
    union { float f; unsigned u; } x;
    x.f = f;
    unsigned r = x.u + 0x7FFFu + ((x.u >> 16) & 1u);   // round-to-nearest-even
    return (unsigned short)(r >> 16);
}

static __device__ inline v16bf load_frag(const unsigned short* p) {
    Frag16 f;
    const uint4* u = (const uint4*)p;
    f.u4[0] = u[0];
    f.u4[1] = u[1];
    return f.v;
}

// Async memory->LDS copy of 16 bytes (gfx1250 GLOBAL_LOAD_ASYNC_TO_LDS_B128,
// tracked by ASYNCcnt; no VGPR data path).
static __device__ inline void async_copy_b128(const void* gptr, void* lds_ptr) {
    unsigned lds_off = (unsigned)(size_t)lds_ptr;   // low 32 bits = LDS offset
    unsigned long long ga = (unsigned long long)gptr;
    asm volatile("global_load_async_to_lds_b128 %0, %1, off"
                 :: "v"(lds_off), "v"(ga)
                 : "memory");
}

static __device__ inline void wait_asynccnt0() {
#if __has_builtin(__builtin_amdgcn_s_wait_asynccnt)
    __builtin_amdgcn_s_wait_asynccnt(0);
#else
    asm volatile("s_wait_asynccnt 0x0" ::: "memory");
#endif
}

// ---------------------------------------------------------------------------
// Elementwise f32 -> bf16 conversion
// ---------------------------------------------------------------------------
__global__ __launch_bounds__(256) void cvt_bf16(const float* __restrict__ x,
                                                unsigned short* __restrict__ y,
                                                int n) {
    int i = blockIdx.x * blockDim.x + threadIdx.x;
    if (i < n) y[i] = f32_to_bf16(x[i]);
}

// ---------------------------------------------------------------------------
// Y[M,N] = Xbf16[M,K] @ Wbf16[N,K]^T   (torch Linear layout, y = x @ W.T)
// Block: 128 threads = 4 waves. Block tile 128x64, wave tile 32x64.
// K-chunks of 32 staged through LDS; 8 WMMAs per wave per chunk
// (2 A-frags + 4 B-frags -> better LDS-load : WMMA ratio).
// ---------------------------------------------------------------------------
template <bool OUT_F32>
__global__ __launch_bounds__(128) void gemm_bf16(const unsigned short* __restrict__ X,
                                                 const unsigned short* __restrict__ W,
                                                 void* __restrict__ Yv,
                                                 int M, int N, int K) {
    __shared__ unsigned short As[128][32];   // 8 KB
    __shared__ unsigned short Bs[64][32];    // 4 KB

    const int t    = threadIdx.x;
    const int wave = t >> 5;
    const int lane = t & 31;
    const int half = lane >> 4;      // 0: K 0..15, 1: K 16..31 of the chunk
    const int l16  = lane & 15;      // row (A) / col (B) within 16

    const int bm = blockIdx.x * 128;
    const int bn = blockIdx.y * 64;

    // A loader: 4 threads fill one 64B LDS row (coalesced global segments)
    const int aRow = t >> 2;         // 0..31 (+32 strides)
    const int aCol = (t & 3) * 8;    // u16 units
    // B loader: 2 threads per row
    const int bRow = t >> 1;         // 0..63
    const int bCol = (t & 1) * 16;

    v8f acc[2][4] = {};

    for (int k0 = 0; k0 < K; k0 += 32) {
        __syncthreads();
#pragma unroll
        for (int rr = 0; rr < 128; rr += 32) {
            *(uint4*)&As[rr + aRow][aCol] =
                *(const uint4*)(X + (size_t)(bm + rr + aRow) * K + k0 + aCol);
        }
        *(uint4*)&Bs[bRow][bCol]     = *(const uint4*)(W + (size_t)(bn + bRow) * K + k0 + bCol);
        *(uint4*)&Bs[bRow][bCol + 8] = *(const uint4*)(W + (size_t)(bn + bRow) * K + k0 + bCol + 8);
        if (k0 + 32 < K) {  // warm caches for the next chunk (global_prefetch_b8)
            __builtin_prefetch(X + (size_t)(bm + aRow) * K + k0 + 32 + aCol, 0, 1);
            __builtin_prefetch(W + (size_t)(bn + bRow) * K + k0 + 32 + bCol, 0, 1);
        }
        __syncthreads();

        v16bf a0 = load_frag(&As[wave * 32 + l16][half * 16]);
        v16bf a1 = load_frag(&As[wave * 32 + 16 + l16][half * 16]);
#pragma unroll
        for (int ct = 0; ct < 4; ++ct) {
            v16bf b = load_frag(&Bs[ct * 16 + l16][half * 16]);
            acc[0][ct] = __builtin_amdgcn_wmma_f32_16x16x32_bf16(
                false, a0, false, b, (short)0, acc[0][ct], false, false);
            acc[1][ct] = __builtin_amdgcn_wmma_f32_16x16x32_bf16(
                false, a1, false, b, (short)0, acc[1][ct], false, false);
        }
    }

#pragma unroll
    for (int s = 0; s < 2; ++s) {
        const int rowb = bm + wave * 32 + s * 16;
#pragma unroll
        for (int ct = 0; ct < 4; ++ct) {
#pragma unroll
            for (int r = 0; r < 8; ++r) {
                int m = r + 8 * half;                 // C layout: row = r + 8*(lane/16)
                int n = bn + ct * 16 + l16;           // col = lane%16
                float vv = acc[s][ct][r];
                if (OUT_F32)
                    ((float*)Yv)[(size_t)(rowb + m) * N + n] = vv;
                else
                    ((unsigned short*)Yv)[(size_t)(rowb + m) * N + n] = f32_to_bf16(vv);
            }
        }
    }
}

// ---------------------------------------------------------------------------
// Flash attention. Block: 256 threads = 8 waves. One block handles 8 query
// tiles (16 rows each) of one (b,h). Iterates key tiles of 32: the K tile is
// staged with GLOBAL_LOAD_ASYNC_TO_LDS_B128 (ASYNCcnt path, no VGPR bounce);
// the V tile is transposed through VGPRs into LDS. Per iteration per wave:
// 4 WMMAs for scores (16x32 over DK=64) + 4 WMMAs for P@V.
// ---------------------------------------------------------------------------
__global__ __launch_bounds__(256) void attn(const unsigned short* __restrict__ Qp,
                                            const unsigned short* __restrict__ Kp,
                                            const unsigned short* __restrict__ Vp,
                                            const int* __restrict__ mask,
                                            unsigned short* __restrict__ Xo) {
    __shared__ unsigned short Ks[32][64];        // [key][d]
    __shared__ unsigned short Vst[64][32];       // [d][key] (transposed)
    __shared__ unsigned short Pst[8][16][32];    // per-wave P staging (bf16)

    const int t    = threadIdx.x;
    const int wave = t >> 5;
    const int lane = t & 31;
    const int half = lane >> 4;
    const int l16  = lane & 15;

    const int bh = blockIdx.x >> 3;      // 0..63  (b*H + h)
    const int qg = blockIdx.x & 7;
    const int b  = bh / H_;
    const int h  = bh % H_;
    const int qtile = qg * 8 + wave;     // 0..63
    const size_t tokQ = (size_t)b * S_ + qtile * 16;

    // staging indices: 2048 u16 per tile / 256 threads = 8 u16 (16B) each
    const int sKey = (t * 8) >> 6;       // 0..31
    const int sD   = (t * 8) & 63;       // 0..56 step 8

    // Q fragments: two K-chunks of 32 covering DK=64
    v16bf qf[2];
#pragma unroll
    for (int c = 0; c < 2; ++c)
        qf[c] = load_frag(Qp + (tokQ + l16) * D_ + h * DK_ + c * 32 + half * 16);

    v8f o[4] = {};
    float mrow[8], lrow[8];
#pragma unroll
    for (int r = 0; r < 8; ++r) { mrow[r] = NEGINF; lrow[r] = 0.0f; }

    const int* mbase = mask + ((size_t)b * S_ + qtile * 16) * S_;

    for (int kb = 0; kb < S_; kb += 32) {
        __syncthreads();
        {
            const size_t tokK = (size_t)b * S_ + kb + sKey;
            // K tile: async memory->LDS (gfx1250 ASYNCcnt path)
            async_copy_b128(Kp + tokK * D_ + h * DK_ + sD, &Ks[sKey][sD]);
            // V tile: transpose through VGPRs into LDS
            unsigned short vt[8];
            *(uint4*)vt = *(const uint4*)(Vp + tokK * D_ + h * DK_ + sD);
#pragma unroll
            for (int i = 0; i < 8; ++i) Vst[sD + i][sKey] = vt[i];
        }
        wait_asynccnt0();
        __syncthreads();

        // scores: S = Q @ K^T  (two 16x16 C tiles covering 32 keys)
        v8f sc0 = {}, sc1 = {};
#pragma unroll
        for (int c = 0; c < 2; ++c) {
            v16bf b0 = load_frag(&Ks[l16][c * 32 + half * 16]);
            v16bf b1 = load_frag(&Ks[16 + l16][c * 32 + half * 16]);
            sc0 = __builtin_amdgcn_wmma_f32_16x16x32_bf16(
                false, qf[c], false, b0, (short)0, sc0, false, false);
            sc1 = __builtin_amdgcn_wmma_f32_16x16x32_bf16(
                false, qf[c], false, b1, (short)0, sc1, false, false);
        }

        // scale + mask + online softmax update
#pragma unroll
        for (int r = 0; r < 8; ++r) {
            int m = r + 8 * half;
            const int* mp = mbase + (size_t)m * S_ + kb;
            float s0 = sc0[r] * SCALE;
            float s1 = sc1[r] * SCALE;
            if (mp[l16] == 0)      s0 = -1.0e9f;
            if (mp[l16 + 16] == 0) s1 = -1.0e9f;

            float rm = fmaxf(s0, s1);
#pragma unroll
            for (int off = 8; off; off >>= 1)
                rm = fmaxf(rm, __shfl_xor(rm, off, 16));

            float mnew = fmaxf(mrow[r], rm);
            float al   = __expf(mrow[r] - mnew);
            float p0   = __expf(s0 - mnew);
            float p1   = __expf(s1 - mnew);
            float ps   = p0 + p1;
#pragma unroll
            for (int off = 8; off; off >>= 1)
                ps += __shfl_xor(ps, off, 16);

            lrow[r] = lrow[r] * al + ps;
            mrow[r] = mnew;
#pragma unroll
            for (int ct = 0; ct < 4; ++ct) o[ct][r] *= al;

            Pst[wave][m][l16]      = f32_to_bf16(p0);
            Pst[wave][m][l16 + 16] = f32_to_bf16(p1);
        }

        // re-fragment P (C layout -> A layout) via per-wave LDS (same-wave DS
        // ops are in-order on CDNA5), then O += P @ V
        v16bf pa = load_frag(&Pst[wave][l16][half * 16]);
#pragma unroll
        for (int ct = 0; ct < 4; ++ct) {
            v16bf vb = load_frag(&Vst[ct * 16 + l16][half * 16]);
            o[ct] = __builtin_amdgcn_wmma_f32_16x16x32_bf16(
                false, pa, false, vb, (short)0, o[ct], false, false);
        }
    }

    // normalize and write bf16 attention output [token, D]
#pragma unroll
    for (int ct = 0; ct < 4; ++ct) {
#pragma unroll
        for (int r = 0; r < 8; ++r) {
            int m = r + 8 * half;
            float vv = o[ct][r] / lrow[r];
            Xo[(tokQ + m) * D_ + h * DK_ + ct * 16 + l16] = f32_to_bf16(vv);
        }
    }
}

// ---------------------------------------------------------------------------
// Host-side launch
// ---------------------------------------------------------------------------
extern "C" void kernel_launch(void* const* d_in, const int* in_sizes, int n_in,
                              void* d_out, int out_size, void* d_ws, size_t ws_size,
                              hipStream_t stream) {
    const float* q    = (const float*)d_in[0];
    const float* k    = (const float*)d_in[1];
    const float* v    = (const float*)d_in[2];
    const int*   mask = (const int*)d_in[3];
    const float* Wq   = (const float*)d_in[4];
    const float* Wk   = (const float*)d_in[5];
    const float* Wv   = (const float*)d_in[6];
    const float* Wo   = (const float*)d_in[7];

    char* ws = (char*)d_ws;
    const size_t ACT  = (size_t)M_TOT * D_ * 2;   // 8 MB bf16 activation
    const size_t WSZ  = (size_t)D_ * D_ * 2;      // 2 MB bf16 weight
    unsigned short* xq  = (unsigned short*)(ws);
    unsigned short* xk  = (unsigned short*)(ws + ACT);
    unsigned short* xv  = (unsigned short*)(ws + 2 * ACT);
    unsigned short* wq  = (unsigned short*)(ws + 3 * ACT);
    unsigned short* wk  = (unsigned short*)(ws + 3 * ACT + WSZ);
    unsigned short* wv  = (unsigned short*)(ws + 3 * ACT + 2 * WSZ);
    unsigned short* wo  = (unsigned short*)(ws + 3 * ACT + 3 * WSZ);
    unsigned short* Qp  = (unsigned short*)(ws + 3 * ACT + 4 * WSZ);
    unsigned short* Kp  = (unsigned short*)(ws + 4 * ACT + 4 * WSZ);
    unsigned short* Vp  = (unsigned short*)(ws + 5 * ACT + 4 * WSZ);
    unsigned short* Xat = xq;  // bf16 query region is dead after the Q GEMM

    const int nAct = M_TOT * D_;     // 4194304
    const int nW   = D_ * D_;        // 1048576

    cvt_bf16<<<nAct / 256, 256, 0, stream>>>(q, xq, nAct);
    cvt_bf16<<<nAct / 256, 256, 0, stream>>>(k, xk, nAct);
    cvt_bf16<<<nAct / 256, 256, 0, stream>>>(v, xv, nAct);
    cvt_bf16<<<nW / 256, 256, 0, stream>>>(Wq, wq, nW);
    cvt_bf16<<<nW / 256, 256, 0, stream>>>(Wk, wk, nW);
    cvt_bf16<<<nW / 256, 256, 0, stream>>>(Wv, wv, nW);
    cvt_bf16<<<nW / 256, 256, 0, stream>>>(Wo, wo, nW);

    dim3 gg(M_TOT / 128, D_ / 64);   // 32 x 16 blocks
    gemm_bf16<false><<<gg, 128, 0, stream>>>(xq, wq, Qp, M_TOT, D_, D_);
    gemm_bf16<false><<<gg, 128, 0, stream>>>(xk, wk, Kp, M_TOT, D_, D_);
    gemm_bf16<false><<<gg, 128, 0, stream>>>(xv, wv, Vp, M_TOT, D_, D_);

    attn<<<B_ * H_ * 8, 256, 0, stream>>>(Qp, Kp, Vp, mask, Xat);

    gemm_bf16<true><<<gg, 128, 0, stream>>>(Xat, wo, (float*)d_out, M_TOT, D_, D_);
}
